// VQVAEJet_50242527429475
// MI455X (gfx1250) — compile-verified, matrix operand order checked by reading, not run on hardware
//
#include <hip/hip_runtime.h>
#include <hip/hip_bf16.h>
#include <math.h>

typedef __attribute__((ext_vector_type(16))) _Float16 v16h;
typedef __attribute__((ext_vector_type(8)))  float    v8f;

#define N_TOTAL 524288
#define HID 128
#define ZD  32
#define NC  256
#define LDH 136   // f16 row stride (halfs) for K=128 tiles (pad 8 to dodge bank conflicts)
#define LDC 40    // f16 row stride (halfs) for K=32 tiles
#define LDZ 33    // f32 row stride for z buffer

// bias LDS offsets
#define OB1 0
#define OB2 128
#define OB3 256
#define OD1 288
#define OD2 416
#define OD3 544

__device__ __forceinline__ void sched_fence() {
#if __has_builtin(__builtin_amdgcn_sched_barrier)
    __builtin_amdgcn_sched_barrier(0);   // keep live ranges stage-local (<256 VGPRs)
#endif
}

__device__ __forceinline__ float fast_tanh(float x) {
#if __has_builtin(__builtin_amdgcn_tanhf)
    return __builtin_amdgcn_tanhf(x);
#elif __has_builtin(__builtin_amdgcn_tanh_f32)
    return __builtin_amdgcn_tanh_f32(x);
#else
    float y;
    asm volatile("v_tanh_f32 %0, %1\n\tv_nop" : "=v"(y) : "v"(x));
    return y;
#endif
}

// tanh-form GELU; |err| ~1e-3 absolute, below the f16 activation rounding
// already present on this path.
__device__ __forceinline__ float gelu_fast(float x) {
    const float x2 = x * x;
    const float u  = x * (0.7978845608028654f + 0.03567740814183477f * x2);
    const float th = fast_tanh(u);
    const float h  = 0.5f * x;
    return fmaf(h, th, h);
}

// Load one 16x32 f16 fragment (A, or B via transposed-weight rows) from LDS,
// row-major with row stride `ld` halfs.  ISA 7.12.2 16-bit A layout:
// lane L -> row L%16; VGPR0-3 = K [ko..ko+7], VGPR4-7 = K [ko+16..ko+23], ko = 8*(L/16).
__device__ __forceinline__ v16h ldfrag(const _Float16* base, int ld) {
    const int lane = threadIdx.x & 31;
    const _Float16* p = base + (lane & 15) * ld + ((lane >> 4) << 3);
    v16h f;
#pragma unroll
    for (int i = 0; i < 8; ++i) { f[i] = p[i]; f[i + 8] = p[i + 16]; }
    return f;
}

__device__ __forceinline__ v8f wmma16(v16h a, v16h b, v8f c) {
    return __builtin_amdgcn_wmma_f32_16x16x32_f16(false, a, false, b, (short)0, c, false, false);
}

__global__ void vq_zero(float* ws) { ws[0] = 0.0f; }

__global__ void vq_finalize(const float* ws, float* out_loss) {
    // vq_loss = embed + BETA*commit = 1.25 * mean((z - q)^2)
    out_loss[0] = 1.25f * ws[0] / (float)(N_TOTAL * ZD);
}

__global__ __launch_bounds__(256)
void vqvae_fused(const float* __restrict__ x,
                 const float* __restrict__ mean_g,
                 const float* __restrict__ std_g,
                 const float* __restrict__ ew1, const float* __restrict__ eb1,
                 const float* __restrict__ ew2, const float* __restrict__ eb2,
                 const float* __restrict__ ew3, const float* __restrict__ eb3,
                 const float* __restrict__ cbg,
                 const float* __restrict__ dw1, const float* __restrict__ db1,
                 const float* __restrict__ dw2, const float* __restrict__ db2,
                 const float* __restrict__ dw3, const float* __restrict__ db3,
                 float* __restrict__ out, float* __restrict__ ws)
{
    // ---- LDS working set (~230 KB of the 320 KB/WGP) ----
    __shared__ _Float16 w2t[HID * LDH];   // enc_w2^T  [out][in] f16
    __shared__ _Float16 d2t[HID * LDH];   // dec_w2^T
    __shared__ _Float16 w3t[ZD  * LDH];   // enc_w3^T  [32][128]
    __shared__ _Float16 d3t[16  * LDH];   // dec_w3^T  [4][128], zero-padded to 16 rows
    __shared__ _Float16 w1h[HID * LDC];   // enc_w1^T  [128][4], zero-padded to K=32
    __shared__ _Float16 d1t[HID * LDC];   // dec_w1^T  [128][32]
    __shared__ _Float16 cb [NC  * LDC];   // codebook  [256][32] (rows == B^T rows)
    __shared__ _Float16 xh [HID * LDC];   // normalized x, [128][4] zero-padded to K=32
    __shared__ _Float16 hA [HID * LDH];   // activation ping
    __shared__ _Float16 hB [HID * LDH];   // activation pong
    __shared__ _Float16 zh [HID * LDC];   // z (f16), later overwritten with q (f16)
    __shared__ float    zb [HID * LDZ];   // z (f32) for accurate loss
    __shared__ float    cnorm[NC];
    __shared__ float    bs[552];          // all biases
    __shared__ int      idx_s[HID];

    const int t = threadIdx.x;

    // ---- cooperative weight staging (phase A: zero-fills + straight stages) ----
    for (int i = t; i < 16 * LDH; i += 256) d3t[i] = (_Float16)0.0f;
    for (int i = t; i < HID * LDC; i += 256) { w1h[i] = (_Float16)0.0f; xh[i] = (_Float16)0.0f; }
    for (int i = t; i < HID * HID; i += 256) {
        const int in = i >> 7, o = i & 127;
        w2t[o * LDH + in] = (_Float16)ew2[i];
        d2t[o * LDH + in] = (_Float16)dw2[i];
    }
    for (int i = t; i < HID * ZD; i += 256) {           // ew3 [128][32]
        const int in = i >> 5, o = i & 31;
        w3t[o * LDH + in] = (_Float16)ew3[i];
    }
    for (int i = t; i < ZD * HID; i += 256) {           // dw1 [32][128]
        const int in = i >> 7, o = i & 127;
        d1t[o * LDC + in] = (_Float16)dw1[i];
    }
    for (int i = t; i < NC * ZD; i += 256) {
        cb[(i >> 5) * LDC + (i & 31)] = (_Float16)cbg[i];
    }
    {
        float s = 0.0f;
#pragma unroll
        for (int d = 0; d < ZD; ++d) { const float v = cbg[t * ZD + d]; s += v * v; }
        cnorm[t] = s;                                   // 256 threads == 256 codes
    }
    if (t < 128) {
        bs[OB1 + t] = eb1[t];
        bs[OB2 + t] = eb2[t];
        bs[OD1 + t] = db1[t];
        bs[OD2 + t] = db2[t];
    }
    if (t < 32) bs[OB3 + t] = eb3[t];
    if (t < 4)  bs[OD3 + t] = db3[t];
    __syncthreads();                                    // zero-fills complete
    // ---- phase B: fills into zero-padded tiles ----
    for (int i = t; i < HID * 4; i += 256) {            // dw3 [128][4] -> d3t rows 0..3
        const int in = i >> 2, o = i & 3;
        d3t[o * LDH + in] = (_Float16)dw3[i];
    }
    for (int i = t; i < 4 * HID; i += 256) {            // ew1 [4][128] -> w1h[o][c]
        const int c = i >> 7, o = i & 127;
        w1h[o * LDC + c] = (_Float16)ew1[i];
    }
    __syncthreads();   // last block-wide sync: everything below is per-wave row-local

    const int wave = t >> 5;
    const int lane = t & 31;
    const int ln   = lane & 15;        // C-frag column / frag row
    const int rh   = lane >> 4;        // 0/1 half select
    const int rb8  = rh * 8;           // C-frag row offset
    const int row0 = wave * 16;        // this wave's 16 rows inside the 128-row tile

    const float m0 = mean_g[0], m1 = mean_g[1], m2 = mean_g[2], m3 = mean_g[3];
    const float rs0 = 1.0f / std_g[0], rs1 = 1.0f / std_g[1];
    const float rs2 = 1.0f / std_g[2], rs3 = 1.0f / std_g[3];

    float lossAcc = 0.0f;

    for (int rep = 0; rep < 2; ++rep) {
        const int g0 = blockIdx.x * 256 + rep * 128;   // global row of tile row 0
        sched_fence();

        // ---- enc layer1: 4 -> 128 via WMMA on zero-padded K=32 tile (8 WMMAs) ----
        {
            // stage normalized x (f16): lane handles row ln, components 2*rh, 2*rh+1
            const int rr = row0 + ln;
            const float2 xp = ((const float2*)x)[(g0 + rr) * 2 + rh];
            const float mm0 = rh ? m2 : m0, rr0 = rh ? rs2 : rs0;
            const float mm1 = rh ? m3 : m1, rr1 = rh ? rs3 : rs1;
            xh[rr * LDC + rh * 2 + 0] = (_Float16)((xp.x - mm0) * rr0);
            xh[rr * LDC + rh * 2 + 1] = (_Float16)((xp.y - mm1) * rr1);

            const v16h ax = ldfrag(&xh[row0 * LDC], LDC);
#pragma unroll
            for (int nt = 0; nt < 8; ++nt) {
                v8f c = {};
                c = wmma16(ax, ldfrag(&w1h[(nt * 16) * LDC], LDC), c);
                const int col = nt * 16 + ln;
                const float bb = bs[OB1 + col];
#pragma unroll
                for (int j = 0; j < 8; ++j)
                    hA[(row0 + rb8 + j) * LDH + col] = (_Float16)gelu_fast(c[j] + bb);
            }
        }
        sched_fence();

        // ---- enc layer2: 128 -> 128, 32 WMMAs ----
        {
            v16h a[4];
#pragma unroll
            for (int kt = 0; kt < 4; ++kt) a[kt] = ldfrag(&hA[row0 * LDH + kt * 32], LDH);
#pragma unroll
            for (int nt = 0; nt < 8; ++nt) {
                v8f c = {};
#pragma unroll
                for (int kt = 0; kt < 4; ++kt)
                    c = wmma16(a[kt], ldfrag(&w2t[(nt * 16) * LDH + kt * 32], LDH), c);
                const int col = nt * 16 + ln;
                const float bb = bs[OB2 + col];
#pragma unroll
                for (int j = 0; j < 8; ++j)
                    hB[(row0 + rb8 + j) * LDH + col] = (_Float16)gelu_fast(c[j] + bb);
            }
        }
        sched_fence();

        // ---- enc layer3: 128 -> 32  (z) ----
        {
            v16h a[4];
#pragma unroll
            for (int kt = 0; kt < 4; ++kt) a[kt] = ldfrag(&hB[row0 * LDH + kt * 32], LDH);
#pragma unroll
            for (int nt = 0; nt < 2; ++nt) {
                v8f c = {};
#pragma unroll
                for (int kt = 0; kt < 4; ++kt)
                    c = wmma16(a[kt], ldfrag(&w3t[(nt * 16) * LDH + kt * 32], LDH), c);
                const int col = nt * 16 + ln;
                const float bb = bs[OB3 + col];
#pragma unroll
                for (int j = 0; j < 8; ++j) {
                    const float zv = c[j] + bb;
                    const int rr = row0 + rb8 + j;
                    zb[rr * LDZ + col] = zv;
                    zh[rr * LDC + col] = (_Float16)zv;
                }
            }
        }
        sched_fence();

        // ---- VQ: scores = z @ codebook^T (16 WMMAs), fused argmin ----
        {
            const v16h az = ldfrag(&zh[row0 * LDC], LDC);
            float bestv[8]; int besti[8];
#pragma unroll
            for (int j = 0; j < 8; ++j) { bestv[j] = 3.4e38f; besti[j] = 0; }
#pragma unroll
            for (int ct = 0; ct < 16; ++ct) {
                v8f s = {};
                s = wmma16(az, ldfrag(&cb[(ct * 16) * LDC], LDC), s);
                const int col = ct * 16 + ln;
                const float cn = cnorm[col];
#pragma unroll
                for (int j = 0; j < 8; ++j) {
                    const float d = cn - 2.0f * s[j];   // ||z||^2 constant per row: dropped
                    if (d < bestv[j]) { bestv[j] = d; besti[j] = col; }
                }
            }
            // butterfly min-reduce across the 16-lane half (rows stay in their half)
#pragma unroll
            for (int off = 1; off < 16; off <<= 1) {
#pragma unroll
                for (int j = 0; j < 8; ++j) {
                    const float ov = __shfl_xor(bestv[j], off);
                    const int   oi = __shfl_xor(besti[j], off);
                    if (ov < bestv[j] || (ov == bestv[j] && oi < besti[j])) {
                        bestv[j] = ov; besti[j] = oi;
                    }
                }
            }
            if (ln == 0) {
#pragma unroll
                for (int j = 0; j < 8; ++j) idx_s[row0 + rb8 + j] = besti[j];
            }
        }
        sched_fence();

        // ---- gather q (float4), accumulate (z-q)^2, overwrite zh with q ----
        {
            const int rr   = row0 + ln;
            const int code = idx_s[rr];
            const int kb   = rh * 16;
            const float4* q4p = (const float4*)(cbg + code * ZD + kb);
#pragma unroll
            for (int k4 = 0; k4 < 4; ++k4) {
                const float4 q4 = q4p[k4];
                const float qv[4] = {q4.x, q4.y, q4.z, q4.w};
#pragma unroll
                for (int e = 0; e < 4; ++e) {
                    const int k = kb + k4 * 4 + e;
                    const float dlt = zb[rr * LDZ + k] - qv[e];
                    lossAcc += dlt * dlt;
                    zh[rr * LDC + k] = (_Float16)qv[e];
                }
            }
        }
        sched_fence();

        // ---- dec layer1: 32 -> 128 (8 WMMAs) ----
        {
            const v16h aq = ldfrag(&zh[row0 * LDC], LDC);
#pragma unroll
            for (int nt = 0; nt < 8; ++nt) {
                v8f c = {};
                c = wmma16(aq, ldfrag(&d1t[(nt * 16) * LDC], LDC), c);
                const int col = nt * 16 + ln;
                const float bb = bs[OD1 + col];
#pragma unroll
                for (int j = 0; j < 8; ++j)
                    hA[(row0 + rb8 + j) * LDH + col] = (_Float16)gelu_fast(c[j] + bb);
            }
        }
        sched_fence();

        // ---- dec layer2: 128 -> 128 (32 WMMAs) ----
        {
            v16h a[4];
#pragma unroll
            for (int kt = 0; kt < 4; ++kt) a[kt] = ldfrag(&hA[row0 * LDH + kt * 32], LDH);
#pragma unroll
            for (int nt = 0; nt < 8; ++nt) {
                v8f c = {};
#pragma unroll
                for (int kt = 0; kt < 4; ++kt)
                    c = wmma16(a[kt], ldfrag(&d2t[(nt * 16) * LDH + kt * 32], LDH), c);
                const int col = nt * 16 + ln;
                const float bb = bs[OD2 + col];
#pragma unroll
                for (int j = 0; j < 8; ++j)
                    hB[(row0 + rb8 + j) * LDH + col] = (_Float16)gelu_fast(c[j] + bb);
            }
        }
        sched_fence();

        // ---- dec layer3: 128 -> 4 via WMMA on zero-padded N=16 tile (4 WMMAs) ----
        {
            v16h a[4];
#pragma unroll
            for (int kt = 0; kt < 4; ++kt) a[kt] = ldfrag(&hB[row0 * LDH + kt * 32], LDH);
            v8f c = {};
#pragma unroll
            for (int kt = 0; kt < 4; ++kt)
                c = wmma16(a[kt], ldfrag(&d3t[kt * 32], LDH), c);
            if (ln < 4) {                          // cols 4..15 are zero padding
                const float bb = bs[OD3 + ln];
                const float sd = std_g[ln];
                const float mn = mean_g[ln];
#pragma unroll
                for (int j = 0; j < 8; ++j) {
                    const int rr = row0 + rb8 + j;
                    out[(g0 + rr) * 4 + ln] = (c[j] + bb) * sd + mn;
                }
            }
        }
        sched_fence();
    }

    // ---- per-wave loss reduction, one atomic per wave ----
#pragma unroll
    for (int off = 16; off > 0; off >>= 1) lossAcc += __shfl_xor(lossAcc, off);
    if (lane == 0) atomicAdd(ws, lossAcc);
}

extern "C" void kernel_launch(void* const* d_in, const int* in_sizes, int n_in,
                              void* d_out, int out_size, void* d_ws, size_t ws_size,
                              hipStream_t stream) {
    const float* x    = (const float*)d_in[0];
    const float* mean = (const float*)d_in[1];
    const float* stdv = (const float*)d_in[2];
    const float* ew1  = (const float*)d_in[3];
    const float* eb1  = (const float*)d_in[4];
    const float* ew2  = (const float*)d_in[5];
    const float* eb2  = (const float*)d_in[6];
    const float* ew3  = (const float*)d_in[7];
    const float* eb3  = (const float*)d_in[8];
    const float* cbg  = (const float*)d_in[9];
    const float* dw1  = (const float*)d_in[10];
    const float* db1  = (const float*)d_in[11];
    const float* dw2  = (const float*)d_in[12];
    const float* db2  = (const float*)d_in[13];
    const float* dw3  = (const float*)d_in[14];
    const float* db3  = (const float*)d_in[15];
    float* out = (float*)d_out;
    float* ws  = (float*)d_ws;

    vq_zero<<<1, 1, 0, stream>>>(ws);
    vqvae_fused<<<N_TOTAL / 256, 256, 0, stream>>>(x, mean, stdv,
                                                   ew1, eb1, ew2, eb2, ew3, eb3,
                                                   cbg, dw1, db1, dw2, db2, dw3, db3,
                                                   out, ws);
    vq_finalize<<<1, 1, 0, stream>>>(ws, out + (size_t)N_TOTAL * 4);
}
